// RPN_2087354106130
// MI455X (gfx1250) — matrix operand order repaired and checked by reading, not compiled, not run on hardware
//
#include <hip/hip_runtime.h>
#include <hip/hip_bf16.h>

#define B_  16
#define C_  512
#define H_  64
#define W_  64
#define HW_ 4096
#define A_  9
#define NANCH (HW_*A_)        // 36864
#define PRE_NMS_ 2000
#define POST_NMS_ 1000
#define NMS_T_ 0.7f
#define BBOX_CLIP_ 4.1351666f  // log(1000/16)

typedef __attribute__((ext_vector_type(16))) __bf16 v16bf;
typedef __attribute__((ext_vector_type(8)))  float  v8f;
typedef __attribute__((ext_vector_type(4)))  unsigned int v4u;

union Frag { v4u u[2]; v16bf bf; };

__device__ __forceinline__ unsigned short f2bf(float f) {
  unsigned u = __float_as_uint(f);
  u += 0x7FFFu + ((u >> 16) & 1u);
  return (unsigned short)(u >> 16);
}
__device__ __forceinline__ unsigned ldsaddr(const void* p) {
  // generic LDS pointer: low 32 bits are the LDS byte offset (aperture addressing)
  return (unsigned)(unsigned long long)p;
}
__device__ __forceinline__ void async_b128(unsigned lds, const void* g) {
  asm volatile("global_load_async_to_lds_b128 %0, %1, off"
               :: "v"(lds), "v"((unsigned long long)g) : "memory");
}
#define WAIT_ASYNC_0() asm volatile("s_wait_asynccnt 0" ::: "memory")
#define WAIT_ASYNC_2() asm volatile("s_wait_asynccnt 2" ::: "memory")

// ---------------- precision / layout transforms ----------------
__global__ void prep_w_k(const float* __restrict__ cw, unsigned short* __restrict__ w2) {
  int i = blockIdx.x*blockDim.x + threadIdx.x;
  if (i >= 9*C_*C_) return;
  int c = i % C_;
  int o = (i / C_) % C_;
  int tap = i / (C_*C_);
  int kh = tap/3, kw = tap%3;
  float v = cw[(((size_t)o*C_ + c)*3 + kh)*3 + kw];
  w2[i] = f2bf(v);   // layout [tap][o][c]
}

__global__ void prep_head_k(const float* __restrict__ clsW, const float* __restrict__ clsB,
                            const float* __restrict__ regW, const float* __restrict__ regB,
                            unsigned short* __restrict__ wc, float* __restrict__ wcb) {
  int i = blockIdx.x*blockDim.x + threadIdx.x;
  if (i < 48*C_) {
    int c = i % C_, n = i / C_;
    float v = 0.f;
    if (n < A_) v = clsW[(size_t)n*C_ + c];
    else if (n < 45) v = regW[(size_t)(n-A_)*C_ + c];
    wc[i] = f2bf(v);  // layout [n][c], n padded to 48
  }
  if (i < 48) {
    float v = 0.f;
    if (i < A_) v = clsB[i];
    else if (i < 45) v = regB[i-A_];
    wcb[i] = v;
  }
}

__global__ void prep_feat_k(const float* __restrict__ feat, unsigned short* __restrict__ fN) {
  size_t i = (size_t)blockIdx.x*blockDim.x + threadIdx.x;
  const size_t total = (size_t)B_*C_*HW_;
  if (i >= total) return;
  size_t c = i % C_;
  size_t p = (i / C_) % HW_;
  size_t b = i / ((size_t)C_*HW_);
  fN[i] = f2bf(feat[(b*C_ + c)*HW_ + p]);   // NCHW fp32 -> NHWC bf16
}

// ---------------- 3x3 conv as implicit GEMM via WMMA bf16 ----------------
// grid: B * 64(m-tiles of 64 px) * 8(n-tiles of 64 ch), block 256 = 8 waves
__global__ __launch_bounds__(256) void conv3_wmma(
    const unsigned short* __restrict__ featN,  // [B][HW][C] bf16
    const unsigned short* __restrict__ w2,     // [9][O][C]  bf16
    const float* __restrict__ bias,            // [512]
    unsigned short* __restrict__ xout)         // [B][HW][O] bf16 (post-ReLU)
{
  __shared__ unsigned short As[2][64*32];
  __shared__ unsigned short Bs[2][64*32];
  const int t   = threadIdx.x;
  const int bid = blockIdx.x;
  const int nb  = bid & 7;
  const int mb  = (bid >> 3) & 63;
  const int b   = bid >> 9;
  const int lane = t & 31;
  const int wv   = t >> 5;
  const int msub = wv & 3;
  const int nhalf = wv >> 2;

  const int srow = t >> 2;          // 0..63
  const int sch  = (t & 3) * 8;     // 0,8,16,24 (bf16 elems)
  const int p0 = mb*64 + srow;
  const int h0 = p0 >> 6, w0 = p0 & 63;

  v8f c0 = {}; v8f c1 = {};

  auto stage = [&](int kk) {
    int tap = kk >> 4;
    int cb  = (kk & 15) * 32;
    int dh = tap / 3 - 1, dw = tap % 3 - 1;
    int hh = min(max(h0 + dh, 0), 63);   // clamp (SAME-pad approximated)
    int ww = min(max(w0 + dw, 0), 63);
    int buf = kk & 1;
    const unsigned short* ga = featN + (((size_t)b*HW_ + (size_t)hh*64 + ww)*C_ + cb + sch);
    async_b128(ldsaddr(&As[buf][srow*32 + sch]), ga);
    const unsigned short* gb = w2 + (((size_t)tap*C_ + nb*64 + srow)*C_ + cb + sch);
    async_b128(ldsaddr(&Bs[buf][srow*32 + sch]), gb);
  };

  stage(0);
  for (int kk = 0; kk < 144; ++kk) {   // 9 taps * 16 k-steps of 32
    int buf = kk & 1;
    if (kk < 143) { stage(kk+1); WAIT_ASYNC_2(); }
    else          { WAIT_ASYNC_0(); }
    __syncthreads();

    Frag a, b0, b1;
    const int m = msub*16 + (lane & 15);
    const int khalf = (lane >> 4) * 8;         // A: lanes<16 K0-7/16-23, lanes>=16 K8-15/24-31
    a.u[0] = *(const v4u*)&As[buf][m*32 + khalf];
    a.u[1] = *(const v4u*)&As[buf][m*32 + 16 + khalf];
    const int n0 = nhalf*32 + (lane & 15);
    const int kb = (lane >> 4) * 8;            // B: lanes<16 K0-7/16-23, lanes>=16 K8-15/24-31
    b0.u[0] = *(const v4u*)&Bs[buf][n0*32 + kb];
    b0.u[1] = *(const v4u*)&Bs[buf][n0*32 + kb + 16];
    b1.u[0] = *(const v4u*)&Bs[buf][(n0+16)*32 + kb];
    b1.u[1] = *(const v4u*)&Bs[buf][(n0+16)*32 + kb + 16];

    c0 = __builtin_amdgcn_wmma_f32_16x16x32_bf16(false, a.bf, false, b0.bf, (short)0, c0, false, false);
    c1 = __builtin_amdgcn_wmma_f32_16x16x32_bf16(false, a.bf, false, b1.bf, (short)0, c1, false, false);
    __syncthreads();
  }

  // epilogue: bias + ReLU + bf16 store, per acc layout (VGPR r: M=r / r+8 by lane half)
  const int Mbase = mb*64 + msub*16 + (lane >> 4)*8;
  const int o0 = nb*64 + nhalf*32 + (lane & 15);
  const int o1 = o0 + 16;
  const float bia0 = bias[o0];
  const float bia1 = bias[o1];
  #pragma unroll
  for (int r = 0; r < 8; ++r) {
    int p = Mbase + r;
    float v0 = fmaxf(c0[r] + bia0, 0.f);
    float v1 = fmaxf(c1[r] + bia1, 0.f);
    size_t base = ((size_t)b*HW_ + p)*C_;
    xout[base + o0] = f2bf(v0);
    xout[base + o1] = f2bf(v1);
  }
}

// ---------------- 1x1 heads (cls 9 + reg 36, padded 48) via WMMA ----------------
// grid: B*256 (16 px per block), block 128 = 4 waves (waves 0..2 compute)
__global__ __launch_bounds__(128) void head_wmma(
    const unsigned short* __restrict__ x,   // [B][HW][C] bf16
    const unsigned short* __restrict__ wc,  // [48][C] bf16
    const float* __restrict__ wcb,          // [48]
    float* __restrict__ clsOut,             // [B][HW*A]
    float* __restrict__ regOut)             // [B][HW*A][4]
{
  __shared__ unsigned short Ax[16*32];
  __shared__ unsigned short Bx[48*32];
  const int t = threadIdx.x;
  const int bid = blockIdx.x;
  const int mb = bid & 255;
  const int b  = bid >> 8;
  const int lane = t & 31;
  const int wv   = t >> 5;

  v8f acc = {};
  for (int kk = 0; kk < 16; ++kk) {
    int cb = kk*32;
    if (t < 64) {
      int row = t >> 2, ch = (t & 3)*8;
      const unsigned short* g = x + (((size_t)b*HW_ + mb*16 + row)*C_ + cb + ch);
      async_b128(ldsaddr(&Ax[row*32 + ch]), g);
    }
    for (int u = t; u < 192; u += 128) {
      int row = u >> 2, ch = (u & 3)*8;
      const unsigned short* g = wc + ((size_t)row*C_ + cb + ch);
      async_b128(ldsaddr(&Bx[row*32 + ch]), g);
    }
    WAIT_ASYNC_0();
    __syncthreads();
    if (wv < 3) {
      Frag a, bb;
      int m = lane & 15;
      int khalf = (lane >> 4)*8;
      a.u[0] = *(const v4u*)&Ax[m*32 + khalf];
      a.u[1] = *(const v4u*)&Ax[m*32 + 16 + khalf];
      int n = wv*16 + (lane & 15);
      int kb = (lane >> 4)*8;
      bb.u[0] = *(const v4u*)&Bx[n*32 + kb];
      bb.u[1] = *(const v4u*)&Bx[n*32 + kb + 16];
      acc = __builtin_amdgcn_wmma_f32_16x16x32_bf16(false, a.bf, false, bb.bf, (short)0, acc, false, false);
    }
    __syncthreads();
  }
  if (wv < 3) {
    int N  = wv*16 + (lane & 15);
    int Mb = (lane >> 4)*8;
    float bv = (N < 45) ? wcb[N] : 0.f;
    #pragma unroll
    for (int r = 0; r < 8; ++r) {
      int p = mb*16 + Mb + r;
      float v = acc[r] + bv;
      if (N < A_) {
        clsOut[(size_t)b*NANCH + (size_t)p*A_ + N] = v;
      } else if (N < 45) {
        int q = N - A_;
        int aa = q >> 2, j = q & 3;
        regOut[(((size_t)b*NANCH) + (size_t)p*A_ + aa)*4 + j] = v;
      }
    }
  }
}

// ---------------- anchors + deltas + clip ----------------
__global__ void proposals_k(const float* __restrict__ regOut,
                            const int* __restrict__ shapes,
                            float* __restrict__ prop) {
  int i = blockIdx.x*blockDim.x + threadIdx.x;
  if (i >= B_*NANCH) return;
  int b = i / NANCH;
  int r = i % NANCH;
  int a = r % A_;
  int pix = r / A_;
  int h = pix >> 6, w = pix & 63;
  const float scales[3] = {32.f, 64.f, 128.f};
  const float sqr[3]    = {0.70710678f, 1.0f, 1.41421356f};
  int si = a / 3, ri = a % 3;
  float hs = scales[si]*sqr[ri];
  float ws = scales[si]/sqr[ri];
  float cx = (w + 0.5f)*16.f;
  float cy = (h + 0.5f)*16.f;
  const float* d = regOut + (size_t)i*4;
  float dx = d[0], dy = d[1];
  float dwv = fminf(fmaxf(d[2], -BBOX_CLIP_), BBOX_CLIP_);
  float dhv = fminf(fmaxf(d[3], -BBOX_CLIP_), BBOX_CLIP_);
  float pcx = cx + dx*ws, pcy = cy + dy*hs;
  float pw = ws*__expf(dwv), ph = hs*__expf(dhv);
  float ih = (float)shapes[b*2+0], iw = (float)shapes[b*2+1];
  float* o = prop + (size_t)i*4;
  o[0] = fminf(fmaxf(pcx - 0.5f*pw, 0.f), iw);
  o[1] = fminf(fmaxf(pcy - 0.5f*ph, 0.f), ih);
  o[2] = fminf(fmaxf(pcx + 0.5f*pw, 0.f), iw);
  o[3] = fminf(fmaxf(pcy + 0.5f*ph, 0.f), ih);
}

// ---------------- top-2000 per batch: radix-select + bitonic sort in LDS ----------------
__global__ __launch_bounds__(1024) void topk_k(
    const float* __restrict__ cls, const float* __restrict__ prop,
    int* __restrict__ topIdx, float* __restrict__ topSc, float* __restrict__ topBox)
{
  __shared__ unsigned hist[256];
  __shared__ unsigned sPref, sRem, cntG, cntE;
  __shared__ unsigned long long sbuf[2048];
  const int b = blockIdx.x;
  const int t = threadIdx.x;
  const float* sc = cls + (size_t)b*NANCH;

  unsigned pref = 0, rem = PRE_NMS_;
  for (int pass = 0; pass < 4; ++pass) {
    int shift = 24 - pass*8;
    if (t < 256) hist[t] = 0;
    __syncthreads();
    for (int i = t; i < NANCH; i += 1024) {
      unsigned key = __float_as_uint(sc[i]);
      key = (key & 0x80000000u) ? ~key : (key | 0x80000000u);
      bool part = (pass == 0) || ((key >> (shift + 8)) == pref);
      if (part) atomicAdd(&hist[(key >> shift) & 255u], 1u);
    }
    __syncthreads();
    if (t == 0) {
      unsigned cum = 0;
      for (int bin = 255; bin >= 0; --bin) {
        unsigned c = hist[bin];
        if (cum + c >= rem) { sPref = (pref << 8) | (unsigned)bin; sRem = rem - cum; break; }
        cum += c;
      }
    }
    __syncthreads();
    pref = sPref; rem = sRem;
    __syncthreads();
  }
  const unsigned T = pref;
  const unsigned nGreater = PRE_NMS_ - rem;
  if (t == 0) { cntG = 0; cntE = 0; }
  for (int j = t; j < 2048; j += 1024) sbuf[j] = 0ull;
  __syncthreads();
  for (int i = t; i < NANCH; i += 1024) {
    unsigned key = __float_as_uint(sc[i]);
    key = (key & 0x80000000u) ? ~key : (key | 0x80000000u);
    if (key > T) {
      unsigned pos = atomicAdd(&cntG, 1u);
      if (pos < PRE_NMS_) sbuf[pos] = ((unsigned long long)key << 32) | (unsigned)(~(unsigned)i);
    }
  }
  __syncthreads();
  for (int i = t; i < NANCH; i += 1024) {
    unsigned key = __float_as_uint(sc[i]);
    key = (key & 0x80000000u) ? ~key : (key | 0x80000000u);
    if (key == T) {
      unsigned pos = nGreater + atomicAdd(&cntE, 1u);
      if (pos < PRE_NMS_) sbuf[pos] = ((unsigned long long)key << 32) | (unsigned)(~(unsigned)i);
    }
  }
  __syncthreads();
  for (int k = 2; k <= 2048; k <<= 1) {
    for (int j = k >> 1; j > 0; j >>= 1) {
      #pragma unroll
      for (int half = 0; half < 2; ++half) {
        int i = t + half*1024;
        int ixj = i ^ j;
        if (ixj > i) {
          unsigned long long va = sbuf[i], vb = sbuf[ixj];
          bool descend = ((i & k) == 0);
          if ((va < vb) == descend) { sbuf[i] = vb; sbuf[ixj] = va; }
        }
      }
      __syncthreads();
    }
  }
  const float* pb = prop + (size_t)b*NANCH*4;
  for (int half = 0; half < 2; ++half) {
    int jj = t + half*1024;
    if (jj < PRE_NMS_) {
      unsigned long long v = sbuf[jj];
      unsigned idx = ~(unsigned)(v & 0xffffffffu);
      topIdx[b*2048 + jj] = (int)idx;
      float lg = sc[idx];
      topSc[b*2048 + jj] = 1.f/(1.f + __expf(-lg));
      float* ob = topBox + ((size_t)b*2048 + jj)*4;
      const float* ib = pb + (size_t)idx*4;
      ob[0]=ib[0]; ob[1]=ib[1]; ob[2]=ib[2]; ob[3]=ib[3];
    }
  }
}

// ---------------- NMS: pairwise suppression bitmasks ----------------
__global__ void nmsmask_k(const float* __restrict__ topBox,
                          unsigned long long* __restrict__ mask) {
  int t = blockIdx.x*blockDim.x + threadIdx.x;
  const int total = B_*PRE_NMS_*32;
  if (t >= total) return;
  int cb = t & 31;
  int r  = (t >> 5) % PRE_NMS_;
  int b  = t / (PRE_NMS_*32);
  const float* bi = topBox + ((size_t)b*2048 + r)*4;
  float ax0=bi[0], ay0=bi[1], ax1=bi[2], ay1=bi[3];
  float areaA = (ax1-ax0)*(ay1-ay0);
  unsigned long long m = 0;
  int j0 = cb*64;
  for (int jj = 0; jj < 64; ++jj) {
    int j = j0 + jj;
    if (j >= PRE_NMS_ || j <= r) continue;
    const float* bj = topBox + ((size_t)b*2048 + j)*4;
    float ix0 = fmaxf(ax0,bj[0]), iy0 = fmaxf(ay0,bj[1]);
    float ix1 = fminf(ax1,bj[2]), iy1 = fminf(ay1,bj[3]);
    float iw = fmaxf(ix1-ix0, 0.f), ih2 = fmaxf(iy1-iy0, 0.f);
    float inter = iw*ih2;
    float areaB = (bj[2]-bj[0])*(bj[3]-bj[1]);
    float iou = inter / (areaA + areaB - inter + 1e-9f);
    if (iou > NMS_T_) m |= (1ull << jj);
  }
  mask[((size_t)b*2048 + r)*32 + cb] = m;
}

// greedy scan: one wave32 per batch, lane l owns keep-bits [64l, 64l+64)
__global__ __launch_bounds__(32) void nmsreduce_k(
    const unsigned long long* __restrict__ mask,
    unsigned long long* __restrict__ keepW) {
  int b = blockIdx.x;
  int lane = threadIdx.x;
  unsigned long long keep = 0;
  for (int bit = 0; bit < 64; ++bit)
    if (lane*64 + bit < PRE_NMS_) keep |= (1ull << bit);
  for (int i = 0; i < PRE_NMS_; ++i) {
    int w = i >> 6, bt = i & 63;
    unsigned lo = (unsigned)__shfl((int)(unsigned)(keep & 0xffffffffu), w, 32);
    unsigned hi = (unsigned)__shfl((int)(unsigned)(keep >> 32), w, 32);
    unsigned long long kw = ((unsigned long long)hi << 32) | lo;
    if ((kw >> bt) & 1ull)
      keep &= ~mask[((size_t)b*2048 + i)*32 + lane];
  }
  keepW[b*32 + lane] = keep;
}

// ---------------- compact kept proposals into output ----------------
__global__ __launch_bounds__(256) void emit_k(
    const unsigned long long* __restrict__ keepW,
    const float* __restrict__ topBox, const float* __restrict__ topSc,
    float* __restrict__ outBox, float* __restrict__ outSc)
{
  __shared__ unsigned part[256];
  __shared__ unsigned nk;
  int b = blockIdx.x, t = threadIdx.x;
  bool kf[8]; unsigned cnt = 0;
  #pragma unroll
  for (int j = 0; j < 8; ++j) {
    int i = t*8 + j;
    bool k = (i < PRE_NMS_) && ((keepW[b*32 + (i>>6)] >> (i & 63)) & 1ull);
    kf[j] = k; cnt += k;
  }
  part[t] = cnt;
  __syncthreads();
  if (t == 0) {
    unsigned run = 0;
    for (int i = 0; i < 256; ++i) { unsigned c = part[i]; part[i] = run; run += c; }
    nk = run;
  }
  __syncthreads();
  unsigned pos = part[t];
  #pragma unroll
  for (int j = 0; j < 8; ++j) {
    int i = t*8 + j;
    if (kf[j]) {
      if (pos < POST_NMS_) {
        const float* ib = topBox + ((size_t)b*2048 + i)*4;
        float* ob = outBox + ((size_t)b*POST_NMS_ + pos)*4;
        ob[0]=ib[0]; ob[1]=ib[1]; ob[2]=ib[2]; ob[3]=ib[3];
        outSc[b*POST_NMS_ + pos] = topSc[b*2048 + i];
      }
      pos++;
    }
  }
  unsigned total = nk > POST_NMS_ ? POST_NMS_ : nk;
  for (int s = t; s < POST_NMS_; s += 256) {
    if ((unsigned)s >= total) {
      float* ob = outBox + ((size_t)b*POST_NMS_ + s)*4;
      ob[0]=0.f; ob[1]=0.f; ob[2]=0.f; ob[3]=0.f;
      outSc[b*POST_NMS_ + s] = 0.f;
    }
  }
}

extern "C" void kernel_launch(void* const* d_in, const int* in_sizes, int n_in,
                              void* d_out, int out_size, void* d_ws, size_t ws_size,
                              hipStream_t stream) {
  const float* feat  = (const float*)d_in[0];
  const int*   shp   = (const int*)d_in[1];
  const float* convW = (const float*)d_in[2];
  const float* convB = (const float*)d_in[3];
  const float* clsW  = (const float*)d_in[4];
  const float* clsB  = (const float*)d_in[5];
  const float* regW  = (const float*)d_in[6];
  const float* regB  = (const float*)d_in[7];
  (void)in_sizes; (void)n_in; (void)out_size; (void)ws_size;

  char* ws = (char*)d_ws;
  size_t cur = 0;
  auto alloc = [&](size_t bytes) { void* p = ws + cur; cur += (bytes + 255) & ~(size_t)255; return p; };
  unsigned short* featN = (unsigned short*)alloc((size_t)B_*HW_*C_*2);
  unsigned short* w2    = (unsigned short*)alloc((size_t)9*C_*C_*2);
  unsigned short* wc    = (unsigned short*)alloc((size_t)48*C_*2);
  float* wcb            = (float*)alloc(48*4);
  unsigned short* x     = (unsigned short*)alloc((size_t)B_*HW_*C_*2);
  float* cls            = (float*)alloc((size_t)B_*NANCH*4);
  float* reg            = (float*)alloc((size_t)B_*NANCH*16);
  float* prop           = (float*)alloc((size_t)B_*NANCH*16);
  int* topIdx           = (int*)alloc((size_t)B_*2048*4);
  float* topSc          = (float*)alloc((size_t)B_*2048*4);
  float* topBox         = (float*)alloc((size_t)B_*2048*16);
  unsigned long long* mask = (unsigned long long*)alloc((size_t)B_*2048*32*8);
  unsigned long long* keep = (unsigned long long*)alloc((size_t)B_*32*8);

  prep_w_k<<<(9*C_*C_ + 255)/256, 256, 0, stream>>>(convW, w2);
  prep_head_k<<<(48*C_ + 255)/256, 256, 0, stream>>>(clsW, clsB, regW, regB, wc, wcb);
  size_t nf = (size_t)B_*C_*HW_;
  prep_feat_k<<<(unsigned)((nf + 255)/256), 256, 0, stream>>>(feat, featN);

  conv3_wmma<<<B_*64*8, 256, 0, stream>>>(featN, w2, convB, x);
  head_wmma<<<B_*256, 128, 0, stream>>>(x, wc, wcb, cls, reg);

  proposals_k<<<(B_*NANCH + 255)/256, 256, 0, stream>>>(reg, shp, prop);
  topk_k<<<B_, 1024, 0, stream>>>(cls, prop, topIdx, topSc, topBox);
  nmsmask_k<<<(B_*PRE_NMS_*32 + 255)/256, 256, 0, stream>>>(topBox, mask);
  nmsreduce_k<<<B_, 32, 0, stream>>>(mask, keep);

  float* outBox = (float*)d_out;
  float* outSc  = outBox + (size_t)B_*POST_NMS_*4;
  emit_k<<<B_, 256, 0, stream>>>(keep, topBox, topSc, outBox, outSc);
}
